// OptimizedLowPrecisionBigNet_67869073211658
// MI455X (gfx1250) — compile-verified, matrix operand order checked by reading, not run on hardware
//
#include <hip/hip_runtime.h>
#include <hip/hip_bf16.h>
#include <cstdint>

// ---------------------------------------------------------------------------
// Types
// ---------------------------------------------------------------------------
typedef __attribute__((ext_vector_type(16))) _Float16 v16h;
typedef __attribute__((ext_vector_type(8)))  _Float16 v8h;
typedef __attribute__((ext_vector_type(8)))  float    v8f;

#define DIM        1024
#define BATCH      8192
#define N_LIN      18
#define N_GROUPS   4096
#define GROUP      256
#define HALF_GROUP 128
#define LN_EPS     1e-5f

// ---------------------------------------------------------------------------
// Dequant: int4 (packed) * scale + zero -> fp16 row-major [DIM, DIM] per layer
// Each thread expands 8 packed bytes -> 16 fp16 (one 32B vector store).
// ---------------------------------------------------------------------------
__global__ __launch_bounds__(256) void dequant_kernel(
    const unsigned char* __restrict__ packed,   // [N_LIN, N_GROUPS, HALF_GROUP]
    const _Float16*      __restrict__ params,   // [N_LIN, N_GROUPS, 2] (scale, zero)
    _Float16*            __restrict__ w16)      // [N_LIN, DIM, DIM]
{
    int idx = blockIdx.x * blockDim.x + threadIdx.x;
    const int chunks_per_group = HALF_GROUP / 8;                // 16
    int l   = idx / (N_GROUPS * chunks_per_group);
    int rem = idx - l * (N_GROUPS * chunks_per_group);
    int g   = rem / chunks_per_group;
    int c   = rem - g * chunks_per_group;
    if (l >= N_LIN) return;

    const unsigned long long* p =
        (const unsigned long long*)(packed + ((size_t)(l * N_GROUPS + g) * HALF_GROUP + c * 8));
    unsigned long long v = *p;

    float s = (float)params[(size_t)(l * N_GROUPS + g) * 2 + 0];
    float z = (float)params[(size_t)(l * N_GROUPS + g) * 2 + 1];

    v16h r;
#pragma unroll
    for (int j = 0; j < 8; ++j) {
        unsigned int byte = (unsigned int)((v >> (8 * j)) & 0xFFu);
        float lo = (float)(byte & 15u);
        float hi = (float)(byte >> 4);
        r[2 * j]     = (_Float16)(lo * s + z);
        r[2 * j + 1] = (_Float16)(hi * s + z);
    }

    int row = g >> 2;
    int col = (g & 3) * GROUP + c * 16;
    _Float16* outp = w16 + (size_t)l * DIM * DIM + (size_t)row * DIM + col;
    *(v16h*)outp = r;
}

// ---------------------------------------------------------------------------
// f32 -> fp16 conversion (initial activation)
// ---------------------------------------------------------------------------
__global__ __launch_bounds__(256) void cvt_f32_to_f16_kernel(
    const float* __restrict__ in, _Float16* __restrict__ out, int n)
{
    int i = (blockIdx.x * blockDim.x + threadIdx.x) * 4;
    if (i + 3 < n) {
#pragma unroll
        for (int j = 0; j < 4; ++j) out[i + j] = (_Float16)in[i + j];
    }
}

// ---------------------------------------------------------------------------
// WMMA GEMM:  C[M,N] = A[M,K] * W[N,K]^T    (K = N = 1024, M = 8192)
// Wave tile 64x64 (4x4 WMMA tiles, 128 accumulator VGPRs).
// Workgroup = 8 waves arranged 2(M) x 4(N) -> 128 x 256 tile.
// Per K-step: 8 fragments (16 b128 loads) feed 16 WMMAs (1:1 ratio).
// A/B fragments loaded straight from row-major memory as 2 x b128 per lane,
// matching the CDNA5 16-bit A/B VGPR layouts (lanes<16: K+0/16; >=16: K+8/24).
// EPI: 0 = ReLU -> fp16, 1 = +residual -> fp16, 2 = +residual -> f32
// ---------------------------------------------------------------------------
__device__ __forceinline__ v16h load_frag(const _Float16* __restrict__ p)
{
    v8h lo = *(const v8h*)p;            // K-chunk 0 (or 8)
    v8h hi = *(const v8h*)(p + 16);     // K-chunk 16 (or 24)
    return __builtin_shufflevector(lo, hi, 0, 1, 2, 3, 4, 5, 6, 7,
                                           8, 9, 10, 11, 12, 13, 14, 15);
}

template <int EPI>
__global__ __launch_bounds__(256) void gemm_wmma_kernel(
    const _Float16* __restrict__ A,      // [BATCH, DIM]
    const _Float16* __restrict__ W,      // [DIM, DIM] fp16 (row = out feature)
    const _Float16* __restrict__ res,    // residual [BATCH, DIM] (EPI 1,2)
    _Float16*       __restrict__ out16,  // EPI 0,1
    float*          __restrict__ out32)  // EPI 2
{
    const int lane  = threadIdx.x & 31;
    const int wave  = threadIdx.x >> 5;
    const int waveM = wave & 1;          // 2 waves along M
    const int waveN = wave >> 1;         // 4 waves along N
    const int m0 = blockIdx.y * 128 + waveM * 64;
    const int n0 = blockIdx.x * 256 + waveN * 64;

    const int lhalf = lane & 15;
    const int koff  = (lane >> 4) << 3;  // lanes 0-15 -> K+0/16, lanes 16-31 -> K+8/24

    const _Float16* aBase = A + (size_t)(m0 + lhalf) * DIM + koff;
    const _Float16* bBase = W + (size_t)(n0 + lhalf) * DIM + koff;

    v8f acc[4][4] = {};

    for (int k0 = 0; k0 < DIM; k0 += 32) {
        v16h af[4];
        v16h bf[4];
#pragma unroll
        for (int mi = 0; mi < 4; ++mi)
            af[mi] = load_frag(aBase + (size_t)(mi * 16) * DIM + k0);
#pragma unroll
        for (int ni = 0; ni < 4; ++ni)
            bf[ni] = load_frag(bBase + (size_t)(ni * 16) * DIM + k0);

#pragma unroll
        for (int mi = 0; mi < 4; ++mi)
#pragma unroll
            for (int ni = 0; ni < 4; ++ni)
                acc[mi][ni] = __builtin_amdgcn_wmma_f32_16x16x32_f16(
                    false, af[mi], false, bf[ni],
                    (short)0, acc[mi][ni], false, false);
    }

    // C/D layout: VGPR i -> row i (lanes 0-15) or i+8 (lanes 16-31); col = lane&15
    const int rOff  = (lane >> 4) << 3;
    const int nLane = lane & 15;
#pragma unroll
    for (int mi = 0; mi < 4; ++mi) {
#pragma unroll
        for (int ni = 0; ni < 4; ++ni) {
#pragma unroll
            for (int i = 0; i < 8; ++i) {
                int m = m0 + mi * 16 + rOff + i;
                int n = n0 + ni * 16 + nLane;
                size_t idx = (size_t)m * DIM + n;
                float v = acc[mi][ni][i];
                if constexpr (EPI == 0) {
                    v = v > 0.0f ? v : 0.0f;
                    out16[idx] = (_Float16)v;
                } else if constexpr (EPI == 1) {
                    out16[idx] = (_Float16)(v + (float)res[idx]);
                } else {
                    out32[idx] = v + (float)res[idx];
                }
            }
        }
    }
}

// ---------------------------------------------------------------------------
// Per-row LayerNorm over DIM=1024 fp16 values. One 256-thread block per row.
// ---------------------------------------------------------------------------
__global__ __launch_bounds__(256) void layernorm_kernel(
    const _Float16* __restrict__ x,
    const _Float16* __restrict__ w,
    const _Float16* __restrict__ b,
    _Float16*       __restrict__ out)
{
    const int row = blockIdx.x;
    const int tid = threadIdx.x;
    const _Float16* p = x + (size_t)row * DIM + tid * 4;

    float v0 = (float)p[0], v1 = (float)p[1], v2 = (float)p[2], v3 = (float)p[3];

    __shared__ float ssum[256];
    __shared__ float ssq[256];
    ssum[tid] = v0 + v1 + v2 + v3;
    ssq[tid]  = v0 * v0 + v1 * v1 + v2 * v2 + v3 * v3;
    __syncthreads();
#pragma unroll
    for (int off = 128; off > 0; off >>= 1) {
        if (tid < off) {
            ssum[tid] += ssum[tid + off];
            ssq[tid]  += ssq[tid + off];
        }
        __syncthreads();
    }
    const float mu   = ssum[0] * (1.0f / DIM);
    const float var  = ssq[0] * (1.0f / DIM) - mu * mu;
    const float rstd = rsqrtf(var + LN_EPS);

    _Float16* q = out + (size_t)row * DIM + tid * 4;
    float vv[4] = {v0, v1, v2, v3};
#pragma unroll
    for (int j = 0; j < 4; ++j) {
        int col = tid * 4 + j;
        float xn = (vv[j] - mu) * rstd;
        q[j] = (_Float16)(xn * (float)w[col] + (float)b[col]);
    }
}

// ---------------------------------------------------------------------------
// Launch
// ---------------------------------------------------------------------------
extern "C" void kernel_launch(void* const* d_in, const int* in_sizes, int n_in,
                              void* d_out, int out_size, void* d_ws, size_t ws_size,
                              hipStream_t stream)
{
    const float*         x_f32    = (const float*)d_in[0];
    const unsigned char* w_packed = (const unsigned char*)d_in[1];
    const _Float16*      w_params = (const _Float16*)d_in[2];
    const _Float16*      ln_w     = (const _Float16*)d_in[3];
    const _Float16*      ln_b     = (const _Float16*)d_in[4];
    float*               out      = (float*)d_out;

    // Workspace layout (fp16): 18 weight matrices + 3 activation buffers
    _Float16* W16  = (_Float16*)d_ws;                       // 18 * 1M halves
    _Float16* act0 = W16  + (size_t)N_LIN * DIM * DIM;      // x / residual
    _Float16* act1 = act0 + (size_t)BATCH * DIM;            // h1 / h3
    _Float16* act2 = act1 + (size_t)BATCH * DIM;            // h2

    // 1) dequantize all 18 layers' weights to fp16 (L2-resident afterwards)
    {
        int total = N_LIN * N_GROUPS * (HALF_GROUP / 8);    // 1,179,648 threads
        dequant_kernel<<<(total + 255) / 256, 256, 0, stream>>>(w_packed, w_params, W16);
    }

    // 2) x (f32) -> fp16
    {
        int n = BATCH * DIM;
        cvt_f32_to_f16_kernel<<<n / (256 * 4), 256, 0, stream>>>(x_f32, act0, n);
    }

    dim3 gGrid(DIM / 256, BATCH / 128);   // (4, 64)
    dim3 gBlk(256);

    // 3) 6 residual blocks
    for (int b = 0; b < 6; ++b) {
        const _Float16* W0 = W16 + (size_t)(b * 3 + 0) * DIM * DIM;
        const _Float16* W1 = W16 + (size_t)(b * 3 + 1) * DIM * DIM;
        const _Float16* W2 = W16 + (size_t)(b * 3 + 2) * DIM * DIM;

        gemm_wmma_kernel<0><<<gGrid, gBlk, 0, stream>>>(act0, W0, nullptr, act1, nullptr);
        gemm_wmma_kernel<0><<<gGrid, gBlk, 0, stream>>>(act1, W1, nullptr, act2, nullptr);

        if (b < 5) {
            // h3 = h2 @ W2^T + x  (fp16), then layernorm -> new x in act0
            gemm_wmma_kernel<1><<<gGrid, gBlk, 0, stream>>>(act2, W2, act0, act1, nullptr);
            layernorm_kernel<<<BATCH, 256, 0, stream>>>(act1, ln_w + b * DIM, ln_b + b * DIM, act0);
        } else {
            // final: write f32 straight to d_out
            gemm_wmma_kernel<2><<<gGrid, gBlk, 0, stream>>>(act2, W2, act0, nullptr, out);
        }
    }
}
